// SlotAttentionExplicitStats_22428319220322
// MI455X (gfx1250) — matrix operand order repaired and checked by reading, not compiled, not run on hardware
//
#include <hip/hip_runtime.h>

// SlotAttentionExplicitStats for MI455X (gfx1250, wave32, WMMA).
// B=64, K=4096, D=64, Q=11, H=128, 3 iterations.
// Memory-bound: f16 k / vT (64MB) stay L2-resident (192MB L2); all recurrent
// per-batch state lives in LDS; GEMMs use v_wmma_f32_16x16x32_f16 with
// fragment-friendly (contiguous 32B per lane) data layouts.

#define BB 64
#define KK 4096
#define DD 64
#define QQ 11
#define HH 128
#define NITER 3
#define EPSF 1e-8f

typedef __attribute__((ext_vector_type(16))) _Float16 v16h;
typedef __attribute__((ext_vector_type(8)))  _Float16 v8h;
typedef __attribute__((ext_vector_type(8)))  float    v8f;

static __device__ __forceinline__ v8f wmma16(v16h a, v16h b, v8f c) {
  // D = A(16x32 f16) * B(32x16 f16) + C(16x16 f32)
  return __builtin_amdgcn_wmma_f32_16x16x32_f16(false, a, false, b,
                                                (short)0, c, false, false);
}

// A-fragment (16x32 f16): per lane, elements 0..15 <- K=half*16+e
// => one contiguous 32B chunk of the source row.
static __device__ __forceinline__ v16h load_afrag(const _Float16* rowp,
                                                  int half) {
  union { v16h v; v8h h[2]; } u;
  u.h[0] = *(const v8h*)(rowp + half * 16);
  u.h[1] = *(const v8h*)(rowp + half * 16 + 8);
  return u.v;
}

// B-fragment (32x16 f16) from a source already contiguous in K for this lane:
// elements e=0..15 <- p[e]  => one 32B chunk.
static __device__ __forceinline__ v16h load_bfrag(const _Float16* p) {
  union { v16h v; v8h h[2]; } u;
  u.h[0] = *(const v8h*)(p);
  u.h[1] = *(const v8h*)(p + 8);
  return u.v;
}

// ---------------------------------------------------------------------------
// Kernel 1: x = LN(LN(feats + grid@W_pe + b_pe)) ; k = x@Wk (row-major f16) ;
// v = x@Wv stored TRANSPOSED as vT[b][d][key] (f16).
// One block = 64 rows of the flattened (B*K, D) problem. 8 waves:
// waves 0-3 produce k for row-tiles 0-3, waves 4-7 produce v.
// Weights are pre-swizzled in LDS into per-lane B-fragment order so each
// fragment load is 2x ds_load_b128.
// ---------------------------------------------------------------------------
__global__ __launch_bounds__(256) void prep_kv_kernel(
    const float* __restrict__ inputs,
    const float* __restrict__ W_pe, const float* __restrict__ b_pe,
    const float* __restrict__ ge_s, const float* __restrict__ ge_b,
    const float* __restrict__ in_s, const float* __restrict__ in_b,
    const float* __restrict__ Wk,  const float* __restrict__ Wv,
    _Float16* __restrict__ kws, _Float16* __restrict__ vwsT)
{
  __shared__ float    xs[64][65];      // f32 staging, stride 65: conflict-free
  __shared__ _Float16 xh[64][64];      // normalized rows (A matrix, f16)
  // Pre-swizzled weights: [dt*4+nt][lane][e], fragment value
  //   B[K=half*16+e][N=nt*16+l16] with lane = half*16+l16, dt = K/32.
  __shared__ _Float16 wk_sw[8][32][16];
  __shared__ _Float16 wv_sw[8][32][16];
  __shared__ float    pe_s[2][64];
  __shared__ float    bpe[64], ges[64], geb[64], ins[64], inb[64];

  const int tid = threadIdx.x;
  for (int i = tid; i < 64 * 64; i += 256) {
    const int din = i >> 6, dout = i & 63;
    const int dt = din >> 5, rem = din & 31;
    const int hf = rem >> 4, e = rem & 15;
    const int nt = dout >> 4, l = dout & 15;
    const int ln = hf * 16 + l;
    wk_sw[dt * 4 + nt][ln][e] = (_Float16)Wk[i];
    wv_sw[dt * 4 + nt][ln][e] = (_Float16)Wv[i];
  }
  if (tid < 128) pe_s[tid >> 6][tid & 63] = W_pe[tid];
  if (tid < 64) {
    bpe[tid] = b_pe[tid];
    ges[tid] = ge_s[tid]; geb[tid] = ge_b[tid];
    ins[tid] = in_s[tid]; inb[tid] = in_b[tid];
  }
  __syncthreads();

  const size_t row0 = (size_t)blockIdx.x * 64;
  {
    const int r = tid >> 2, seg = tid & 3;       // 4 threads per row
    const float* ip = inputs + (row0 + r) * 66;
    const float g0 = ip[64], g1 = ip[65];
#pragma unroll
    for (int j = 0; j < 16; ++j) {
      const int d = seg * 16 + j;
      xs[r][d] = ip[d] + g0 * pe_s[0][d] + g1 * pe_s[1][d] + bpe[d];
    }
  }
  __syncthreads();

  if (tid < 64) {                                 // per-row double LayerNorm
    float m = 0.f;
    for (int d = 0; d < 64; ++d) m += xs[tid][d];
    m *= (1.f / 64.f);
    float v = 0.f;
    for (int d = 0; d < 64; ++d) { float t = xs[tid][d] - m; v += t * t; }
    v *= (1.f / 64.f);
    float inv = rsqrtf(v + 1e-6f);
    for (int d = 0; d < 64; ++d)
      xs[tid][d] = (xs[tid][d] - m) * inv * ges[d] + geb[d];
    m = 0.f;
    for (int d = 0; d < 64; ++d) m += xs[tid][d];
    m *= (1.f / 64.f);
    v = 0.f;
    for (int d = 0; d < 64; ++d) { float t = xs[tid][d] - m; v += t * t; }
    v *= (1.f / 64.f);
    inv = rsqrtf(v + 1e-6f);
    for (int d = 0; d < 64; ++d)
      xh[tid][d] = (_Float16)((xs[tid][d] - m) * inv * ins[d] + inb[d]);
  }
  __syncthreads();

  const int wave = tid >> 5, lane = tid & 31;
  const int half = lane >> 4, l16 = lane & 15;
  const int rt    = wave & 3;     // 16-row tile index
  const int which = wave >> 2;    // 0 -> k, 1 -> v (transposed store)

  const _Float16* rowp = &xh[rt * 16 + l16][0];
  const v16h a0 = load_afrag(rowp, half);        // K 0..31
  const v16h a1 = load_afrag(rowp + 32, half);   // K 32..63
  const _Float16* wsw = which ? &wv_sw[0][0][0] : &wk_sw[0][0][0];

#pragma unroll
  for (int nt = 0; nt < 4; ++nt) {
    const v16h b0 = load_bfrag(wsw + ((0 * 4 + nt) * 32 + lane) * 16);
    const v16h b1 = load_bfrag(wsw + ((1 * 4 + nt) * 32 + lane) * 16);
    v8f acc = {};
    acc = wmma16(a1, b1, acc);
    acc = wmma16(a0, b0, acc);
    if (which) {
      // vT[b][d][key]: lane owns 8 consecutive keys -> one 16B store
      const size_t bb   = row0 >> 12;                       // row0 / K
      const size_t key0 = (row0 & (KK - 1)) + rt * 16 + half * 8;
      v8h pk;
#pragma unroll
      for (int i = 0; i < 8; ++i) pk[i] = (_Float16)acc[i];
      *(v8h*)&vwsT[((bb * 64) + nt * 16 + l16) * (size_t)KK + key0] = pk;
    } else {
      // k row-major [row][d]
#pragma unroll
      for (int i = 0; i < 8; ++i)
        kws[(row0 + rt * 16 + half * 8 + i) * 64 + nt * 16 + l16] =
            (_Float16)acc[i];
    }
  }
}

// ---------------------------------------------------------------------------
// Kernel 2: 3 fused slot-attention iterations + output stats.
// One workgroup per batch, 16 waves. Attention is streamed: per 16x16 logits
// tile -> in-register softmax over the Q axis (lane pair n/n+16 via shfl_xor),
// deferred 1/(sum_k+eps) normalization, a@v on WMMA (vT gives contiguous
// B-fragments) via per-wave LDS staging of the a tile.
// ---------------------------------------------------------------------------
__global__ __launch_bounds__(512) void slot_iter_kernel(
    const float* __restrict__ slots_in, const float* __restrict__ inputs,
    const float* __restrict__ q_s, const float* __restrict__ q_b,
    const float* __restrict__ Wq,
    const float* __restrict__ Wir, const float* __restrict__ Wiz,
    const float* __restrict__ Win,
    const float* __restrict__ bir, const float* __restrict__ biz,
    const float* __restrict__ bin_,
    const float* __restrict__ Whr, const float* __restrict__ Whz,
    const float* __restrict__ Whn, const float* __restrict__ bhn,
    const float* __restrict__ mlp_s, const float* __restrict__ mlp_b,
    const float* __restrict__ W1, const float* __restrict__ b1,
    const float* __restrict__ W2, const float* __restrict__ b2,
    const _Float16* __restrict__ kws, const _Float16* __restrict__ vwsT,
    _Float16* __restrict__ aws, float* __restrict__ out)
{
  __shared__ float    slots_s[QQ][64];
  __shared__ float    tslots[QQ][64];
  __shared__ float    lnm[QQ][64];
  __shared__ _Float16 qtile[16][64];       // padded q (rows 11..15 zero)
  __shared__ _Float16 stage[16][16][32];   // per-wave 16x32 attn A-tile staging
  __shared__ float    upd[16][64];
  __shared__ float    rowsum[16];
  __shared__ float    hbuf[QQ][HH];
  __shared__ float    pos_s[QQ][2];
  __shared__ float    ss_s[QQ][2];

  const int tid  = threadIdx.x;
  const int bidx = blockIdx.x;
  const int wave = tid >> 5, lane = tid & 31;
  const int half = lane >> 4, l16 = lane & 15;
  const float scale = 0.125f;  // 1/sqrt(64)

  for (int i = tid; i < QQ * 64; i += 512)
    slots_s[i >> 6][i & 63] =
        slots_in[((size_t)bidx * QQ + (i >> 6)) * 68 + (i & 63)];
  __syncthreads();

  for (int it = 0; it < NITER; ++it) {
    // (a) LN(slots) with q_s/q_b
    if (tid < QQ) {
      float m = 0.f;
      for (int d = 0; d < 64; ++d) m += slots_s[tid][d];
      m *= (1.f / 64.f);
      float v = 0.f;
      for (int d = 0; d < 64; ++d) { float t = slots_s[tid][d] - m; v += t * t; }
      v *= (1.f / 64.f);
      float inv = rsqrtf(v + 1e-6f);
      for (int d = 0; d < 64; ++d)
        lnm[tid][d] = (slots_s[tid][d] - m) * inv * q_s[d] + q_b[d];
    }
    if (tid < 16) rowsum[tid] = 0.f;
    __syncthreads();

    // (b) q = lnm @ Wq * scale (f16, padded to 16 rows); zero updates accum
    for (int i = tid; i < 16 * 64; i += 512) {
      const int r = i >> 6, d = i & 63;
      upd[r][d] = 0.f;
      float acc = 0.f;
      if (r < QQ) {
        for (int dd = 0; dd < 64; ++dd) acc += lnm[r][dd] * Wq[dd * 64 + d];
        acc *= scale;
      }
      qtile[r][d] = (_Float16)acc;
    }
    __syncthreads();

    // (c) streaming logits -> softmax(Q) -> staged a -> a@v, per 32-key pair
    const v16h qa0 = load_afrag(&qtile[l16][0], half);     // d 0..31
    const v16h qa1 = load_afrag(&qtile[l16][32], half);    // d 32..63
    v8f accn[4] = {};
    float rs[8] = {0.f, 0.f, 0.f, 0.f, 0.f, 0.f, 0.f, 0.f};
    const int last = (it == NITER - 1);

    for (int pr = 0; pr < 8; ++pr) {            // 8 pairs of 16-key tiles/wave
      const int kb = wave * 256 + pr * 32;
#pragma unroll
      for (int sub = 0; sub < 2; ++sub) {
        const int kbase = kb + sub * 16;
        // B[d][key]: k row (contiguous d) -> contiguous 32B per lane
        const _Float16* kp = kws + ((size_t)bidx * KK + kbase + l16) * 64;
        const v16h b0 = load_bfrag(kp + half * 16);
        const v16h b1 = load_bfrag(kp + 32 + half * 16);
        v8f lg = {};
        lg = wmma16(qa1, b1, lg);
        lg = wmma16(qa0, b0, lg);
        // softmax over Q (column dir); lane n holds rows half*8..half*8+7
        float pm = -1e30f;
#pragma unroll
        for (int i = 0; i < 8; ++i)
          if (half * 8 + i < QQ) pm = fmaxf(pm, lg[i]);
        const float cm = fmaxf(pm, __shfl_xor(pm, 16, 32));
        float ev[8];
        float ps = 0.f;
#pragma unroll
        for (int i = 0; i < 8; ++i) {
          ev[i] = (half * 8 + i < QQ) ? __expf(lg[i] - cm) : 0.f;
          ps += ev[i];
        }
        const float inv = 1.f / (ps + __shfl_xor(ps, 16, 32));
#pragma unroll
        for (int i = 0; i < 8; ++i) {
          ev[i] *= inv;
          rs[i] += ev[i];
          stage[wave][half * 8 + i][sub * 16 + l16] = (_Float16)ev[i];
        }
        if (last) {
#pragma unroll
          for (int i = 0; i < 8; ++i)
            if (half * 8 + i < QQ)
              aws[((size_t)bidx * QQ + half * 8 + i) * KK + kbase + l16] =
                  (_Float16)ev[i];
        }
      }
      // a-tile (16 q x 32 keys) as A fragment; same-wave LDS RAW is ordered
      const v16h af = load_afrag(&stage[wave][l16][0], half);
#pragma unroll
      for (int nt = 0; nt < 4; ++nt) {
        // vT[b][d][key]: contiguous 32B per lane in the key (K) dimension
        const v16h vb = load_bfrag(
            vwsT + ((size_t)bidx * 64 + nt * 16 + l16) * (size_t)KK + kb +
            half * 16);
        accn[nt] = wmma16(af, vb, accn[nt]);
      }
    }
#pragma unroll
    for (int i = 0; i < 8; ++i) atomicAdd(&rowsum[half * 8 + i], rs[i]);
#pragma unroll
    for (int nt = 0; nt < 4; ++nt)
#pragma unroll
      for (int i = 0; i < 8; ++i)
        atomicAdd(&upd[half * 8 + i][nt * 16 + l16], accn[nt][i]);
    __syncthreads();

    // (d) deferred attention normalization: updates *= 1/(sum_k a + eps)
    for (int i = tid; i < QQ * 64; i += 512) {
      const int q = i >> 6;
      upd[q][i & 63] *= 1.f / (rowsum[q] + EPSF);
    }
    __syncthreads();

    // (e) GRU
    for (int i = tid; i < QQ * 64; i += 512) {
      const int q = i >> 6, d = i & 63;
      float ar = bir[d], az = biz[d], an = bin_[d], hn = bhn[d];
      float hr = 0.f, hz = 0.f;
      for (int dd = 0; dd < 64; ++dd) {
        const float u = upd[q][dd], s = slots_s[q][dd];
        ar += u * Wir[dd * 64 + d];
        az += u * Wiz[dd * 64 + d];
        an += u * Win[dd * 64 + d];
        hr += s * Whr[dd * 64 + d];
        hz += s * Whz[dd * 64 + d];
        hn += s * Whn[dd * 64 + d];
      }
      const float r  = 1.f / (1.f + __expf(-(ar + hr)));
      const float z  = 1.f / (1.f + __expf(-(az + hz)));
      const float nn = tanhf(an + r * hn);
      tslots[q][d] = (1.f - z) * nn + z * slots_s[q][d];
    }
    __syncthreads();

    // (f) pre-LN MLP residual
    if (tid < QQ) {
      float m = 0.f;
      for (int d = 0; d < 64; ++d) m += tslots[tid][d];
      m *= (1.f / 64.f);
      float v = 0.f;
      for (int d = 0; d < 64; ++d) { float t = tslots[tid][d] - m; v += t * t; }
      v *= (1.f / 64.f);
      float inv = rsqrtf(v + 1e-6f);
      for (int d = 0; d < 64; ++d)
        lnm[tid][d] = (tslots[tid][d] - m) * inv * mlp_s[d] + mlp_b[d];
    }
    __syncthreads();
    for (int i = tid; i < QQ * HH; i += 512) {
      const int q = i >> 7, j = i & 127;
      float acc = b1[j];
      for (int dd = 0; dd < 64; ++dd) acc += lnm[q][dd] * W1[dd * HH + j];
      hbuf[q][j] = fmaxf(acc, 0.f);
    }
    __syncthreads();
    for (int i = tid; i < QQ * 64; i += 512) {
      const int q = i >> 6, d = i & 63;
      float acc = tslots[q][d] + b2[d];
      for (int j = 0; j < HH; ++j) acc += hbuf[q][j] * W2[j * 64 + d];
      slots_s[q][d] = acc;
    }
    __syncthreads();
  }

  // positions = (a @ grid) / (rowsum + eps)
  if (tid < QQ * 2) {
    pos_s[tid >> 1][tid & 1] = 0.f;
    ss_s[tid >> 1][tid & 1]  = 0.f;
  }
  __syncthreads();
  {
    float lp[QQ * 2];
#pragma unroll
    for (int i = 0; i < QQ * 2; ++i) lp[i] = 0.f;
    for (int c = tid; c < KK; c += 512) {
      const float* gp = inputs + ((size_t)bidx * KK + c) * 66 + 64;
      const float g0 = gp[0], g1 = gp[1];
#pragma unroll
      for (int q = 0; q < QQ; ++q) {
        const float a = (float)aws[((size_t)bidx * QQ + q) * KK + c];
        lp[q * 2]     += a * g0;
        lp[q * 2 + 1] += a * g1;
      }
    }
#pragma unroll
    for (int i = 0; i < QQ * 2; ++i) atomicAdd(&pos_s[i >> 1][i & 1], lp[i]);
  }
  __syncthreads();
  if (tid < QQ * 2)
    pos_s[tid >> 1][tid & 1] *= 1.f / (rowsum[tid >> 1] + EPSF);
  __syncthreads();

  // scales = clip(sqrt(sum_k (attn+eps) * (grid-pos)^2), 0.01, 5)
  {
    float sinv[QQ], p0[QQ], p1[QQ];
#pragma unroll
    for (int q = 0; q < QQ; ++q) {
      sinv[q] = 1.f / (rowsum[q] + EPSF);
      p0[q] = pos_s[q][0];
      p1[q] = pos_s[q][1];
    }
    float ls[QQ * 2];
#pragma unroll
    for (int i = 0; i < QQ * 2; ++i) ls[i] = 0.f;
    for (int c = tid; c < KK; c += 512) {
      const float* gp = inputs + ((size_t)bidx * KK + c) * 66 + 64;
      const float g0 = gp[0], g1 = gp[1];
#pragma unroll
      for (int q = 0; q < QQ; ++q) {
        const float attn =
            (float)aws[((size_t)bidx * QQ + q) * KK + c] * sinv[q] + EPSF;
        const float d0 = g0 - p0[q], d1 = g1 - p1[q];
        ls[q * 2]     += attn * d0 * d0;
        ls[q * 2 + 1] += attn * d1 * d1;
      }
    }
#pragma unroll
    for (int i = 0; i < QQ * 2; ++i) atomicAdd(&ss_s[i >> 1][i & 1], ls[i]);
  }
  __syncthreads();

  for (int i = tid; i < QQ * 64; i += 512) {
    const int q = i >> 6, d = i & 63;
    out[((size_t)bidx * QQ + q) * 68 + d] = slots_s[q][d];
  }
  if (tid < QQ * 2) {
    const int q = tid >> 1, j = tid & 1;
    out[((size_t)bidx * QQ + q) * 68 + 64 + j] = pos_s[q][j];
    float sc = sqrtf(ss_s[q][j]);
    sc = fminf(fmaxf(sc, 0.01f), 5.0f);
    out[((size_t)bidx * QQ + q) * 68 + 66 + j] = sc;
  }
}

extern "C" void kernel_launch(void* const* d_in, const int* in_sizes, int n_in,
                              void* d_out, int out_size, void* d_ws,
                              size_t ws_size, hipStream_t stream) {
  const float* slots  = (const float*)d_in[0];
  const float* inputs = (const float*)d_in[1];
  const float* W_pe   = (const float*)d_in[2];
  const float* b_pe   = (const float*)d_in[3];
  const float* ge_s   = (const float*)d_in[4];
  const float* ge_b   = (const float*)d_in[5];
  const float* in_s   = (const float*)d_in[6];
  const float* in_b   = (const float*)d_in[7];
  const float* Wk     = (const float*)d_in[8];
  const float* Wv     = (const float*)d_in[9];
  const float* q_s    = (const float*)d_in[10];
  const float* q_b    = (const float*)d_in[11];
  const float* Wq     = (const float*)d_in[12];
  const float* Wir    = (const float*)d_in[13];
  const float* Wiz    = (const float*)d_in[14];
  const float* Win    = (const float*)d_in[15];
  const float* bir    = (const float*)d_in[16];
  const float* biz    = (const float*)d_in[17];
  const float* bin_   = (const float*)d_in[18];
  const float* Whr    = (const float*)d_in[19];
  const float* Whz    = (const float*)d_in[20];
  const float* Whn    = (const float*)d_in[21];
  const float* bhn    = (const float*)d_in[22];
  const float* mlp_s  = (const float*)d_in[23];
  const float* mlp_b  = (const float*)d_in[24];
  const float* W1     = (const float*)d_in[25];
  const float* b1     = (const float*)d_in[26];
  const float* W2     = (const float*)d_in[27];
  const float* b2     = (const float*)d_in[28];
  float* out = (float*)d_out;

  // Workspace layout (f16): k (32MB) | vT (32MB) | a (5.75MB)  ~= 70MB total.
  _Float16* kws  = (_Float16*)d_ws;
  _Float16* vwsT = kws + (size_t)BB * KK * DD;
  _Float16* aws  = vwsT + (size_t)BB * KK * DD;

  prep_kv_kernel<<<(BB * KK) / 64, 256, 0, stream>>>(
      inputs, W_pe, b_pe, ge_s, ge_b, in_s, in_b, Wk, Wv, kws, vwsT);

  slot_iter_kernel<<<BB, 512, 0, stream>>>(
      slots, inputs, q_s, q_b, Wq, Wir, Wiz, Win, bir, biz, bin_, Whr, Whz,
      Whn, bhn, mlp_s, mlp_b, W1, b1, W2, b2, kws, vwsT, aws, out);
}
